// LLM_21337397526807
// MI455X (gfx1250) — matrix operand order repaired and checked by reading, not compiled, not run on hardware
//
#include <hip/hip_runtime.h>
#include <stdint.h>

// Problem constants (fixed by the reference)
#define BB    4
#define SS    2048
#define MM    (BB * SS)      // 8192 token rows
#define HIDN  1024           // output dim per matrix
#define ENGH  1024           // feature (K) dim
#define NMAT  5              // value + 4 key heads
#define PADID 2

typedef __attribute__((ext_vector_type(16))) __bf16 v16bf;
typedef __attribute__((ext_vector_type(8)))  float  v8f;
typedef __attribute__((ext_vector_type(4)))  int    v4i;

#if __has_builtin(__builtin_amdgcn_global_load_async_to_lds_b128) && \
    __has_builtin(__builtin_amdgcn_s_wait_asynccnt)
#define HAVE_ASYNC_LDS 1
#else
#define HAVE_ASYNC_LDS 0
#endif

#if HAVE_ASYNC_LDS
// 16B global -> LDS async copy (ASYNCcnt tracked, no VGPR data round-trip).
// Builtin signature (from compiler diagnostic): (v4i AS1*, v4i AS3*, imm, imm).
static __device__ __forceinline__ void async_cp16(const void* g, void* l) {
    __builtin_amdgcn_global_load_async_to_lds_b128(
        (v4i __attribute__((address_space(1)))*)g,
        (v4i __attribute__((address_space(3)))*)l,
        0, 0);
}
#endif

static __device__ __forceinline__ unsigned short f32_to_bf16_rne(float x) {
    unsigned u = __float_as_uint(x);
    u += ((u >> 16) & 1u) + 0x7FFFu;   // round-to-nearest-even
    return (unsigned short)(u >> 16);
}

// Split float into bf16 hi + bf16 lo (hi = RNE(x), lo = RNE(x - hi)).
static __device__ __forceinline__ void split_bf16(float x, unsigned short& h, unsigned short& l) {
    h = f32_to_bf16_rne(x);
    float hf = __uint_as_float(((unsigned)h) << 16);
    l = f32_to_bf16_rne(x - hf);
}

// ---------------------------------------------------------------------------
// Kernel 1: n-gram hash gather -> feat stored as bf16 hi/lo planes.
// 1 block = 1 token; 256 threads: thread t -> head t/16, float4 chunk t%16.
// ---------------------------------------------------------------------------
__global__ void gather_feat_kernel(const int* __restrict__ ids,
                                   const int* __restrict__ mult,
                                   const int* __restrict__ vs,
                                   const int* __restrict__ off,
                                   const float* __restrict__ emb,
                                   unsigned short* __restrict__ fhi,
                                   unsigned short* __restrict__ flo) {
    const int tok = blockIdx.x;              // 0..MM-1
    const int b = tok >> 11, s = tok & (SS - 1);
    const int t = threadIdx.x;
    const int head = t >> 4;                 // 0..15
    const int chunk = t & 15;                // float4 within 64-float row

    const long long id0 = ids[(size_t)b * SS + s];
    const long long id1 = (s >= 1) ? (long long)ids[(size_t)b * SS + s - 1] : (long long)PADID;
    const long long id2 = (s >= 2) ? (long long)ids[(size_t)b * SS + s - 2] : (long long)PADID;
    const unsigned long long m0 = (unsigned long long)mult[0];
    const unsigned long long m1 = (unsigned long long)mult[1];
    const unsigned long long m2 = (unsigned long long)mult[2];

    const unsigned long long mix2 = (unsigned long long)id0 * m0 + (unsigned long long)id1 * m1;
    const unsigned long long mix3 = mix2 + (unsigned long long)id2 * m2;
    const unsigned long long mix = (head < 8) ? mix2 : mix3;

    const unsigned long long v = (unsigned long long)vs[head];
    const unsigned long long o = (unsigned long long)off[head];
    const unsigned long long idx = o + (mix % v);

    const float4 e = *(const float4*)(emb + idx * 64ull + (size_t)chunk * 4);

    unsigned short h0, h1, h2, h3, l0, l1, l2, l3;
    split_bf16(e.x, h0, l0);
    split_bf16(e.y, h1, l1);
    split_bf16(e.z, h2, l2);
    split_bf16(e.w, h3, l3);

    uint2 ph, pl;
    ph.x = (unsigned)h0 | ((unsigned)h1 << 16);
    ph.y = (unsigned)h2 | ((unsigned)h3 << 16);
    pl.x = (unsigned)l0 | ((unsigned)l1 << 16);
    pl.y = (unsigned)l2 | ((unsigned)l3 << 16);

    const size_t e0 = (size_t)tok * ENGH + (size_t)head * 64 + (size_t)chunk * 4;
    *(uint2*)(fhi + e0) = ph;
    *(uint2*)(flo + e0) = pl;
}

// ---------------------------------------------------------------------------
// Kernel 2: split the 5 (N=1024 x K=1024) weight matrices into bf16 hi/lo.
// value_w is matrix 0, key_w[c] are matrices 1..4; all row-major (N, K).
// ---------------------------------------------------------------------------
__global__ void wconv_kernel(const float* __restrict__ vw,
                             const float* __restrict__ kw,
                             unsigned short* __restrict__ whi,
                             unsigned short* __restrict__ wlo) {
    const size_t i = ((size_t)blockIdx.x * blockDim.x + threadIdx.x) * 4; // element idx
    const float* src = (i < (size_t)HIDN * ENGH) ? (vw + i)
                                                 : (kw + (i - (size_t)HIDN * ENGH));
    const float4 x = *(const float4*)src;

    unsigned short h0, h1, h2, h3, l0, l1, l2, l3;
    split_bf16(x.x, h0, l0);
    split_bf16(x.y, h1, l1);
    split_bf16(x.z, h2, l2);
    split_bf16(x.w, h3, l3);

    uint2 ph, pl;
    ph.x = (unsigned)h0 | ((unsigned)h1 << 16);
    ph.y = (unsigned)h2 | ((unsigned)h3 << 16);
    pl.x = (unsigned)l0 | ((unsigned)l1 << 16);
    pl.y = (unsigned)l2 | ((unsigned)l3 << 16);

    *(uint2*)(whi + i) = ph;
    *(uint2*)(wlo + i) = pl;
}

// ---------------------------------------------------------------------------
// Kernel 3: GEMM, D[g] = feat (8192x1024) x W[g]^T (N,K row-major) + bias.
// bf16x3 split accumulation via v_wmma_f32_16x16x32_bf16 (wave32).
// Block: 256 threads = 8 waves, tile 128(M) x 64(N); wave tile 32x32.
// LDS rows padded to 40 bf16 (80B) to spread banks; double-buffered with
// GLOBAL_LOAD_ASYNC_TO_LDS_B128 (ASYNCcnt) so DMA of chunk kc+1 overlaps the
// WMMA work on chunk kc.
// ---------------------------------------------------------------------------
#define SMHALF 15360                     // ushorts per K-chunk buffer (30 KB)
#define AHOFF  0
#define ALOFF  5120
#define BHOFF  10240
#define BLOFF  12800

__global__ __launch_bounds__(256)
void gemm_bf16x3_kernel(const unsigned short* __restrict__ fhi,
                        const unsigned short* __restrict__ flo,
                        const unsigned short* __restrict__ whi,
                        const unsigned short* __restrict__ wlo,
                        const float* __restrict__ value_b,
                        const float* __restrict__ key_b,
                        float* __restrict__ out) {
#if HAVE_ASYNC_LDS
    __shared__ unsigned short sm[2 * SMHALF];  // 60 KB, double-buffered
#else
    __shared__ unsigned short sm[SMHALF];      // 30 KB, single buffer
#endif

    const int t = threadIdx.x;
    const int gn0 = blockIdx.x * 64;           // global col across 5*1024
    const int g = gn0 >> 10;                   // which weight matrix
    const int n0 = gn0 & 1023;                 // col within matrix
    const int m0 = blockIdx.y * 128;

    const int lane = t & 31, wave = t >> 5;
    const int waveM = wave >> 1, waveN = wave & 1;
    const int mrow = lane & 15;
    const int hi16 = lane >> 4;                // 0 for lanes 0-15, 1 for 16-31
    const int kbA = hi16 * 16;                 // A: K base (contiguous, matches B)
    const int kbB = hi16 * 16;                 // B: K base (contiguous, matches A)

    const size_t wbase = (size_t)g * ((size_t)HIDN * ENGH);

    v8f acc[2][2] = {};

    // Fragment load + 12 WMMAs for one staged K-chunk.
    auto compute = [&](const unsigned short* smb) {
        union Frag { uint4 q[2]; v16bf v; };
        Frag aH[2], aL[2], bH[2], bL[2];
#pragma unroll
        for (int mi = 0; mi < 2; ++mi) {
            const int m = waveM * 32 + mi * 16 + mrow;
            const int base = m * 40;
            // v0..v3: K = kbA..kbA+7 ; v4..v7: K = kbA+8..kbA+15
            aH[mi].q[0] = *(const uint4*)&smb[AHOFF + base + kbA];
            aH[mi].q[1] = *(const uint4*)&smb[AHOFF + base + kbA + 8];
            aL[mi].q[0] = *(const uint4*)&smb[ALOFF + base + kbA];
            aL[mi].q[1] = *(const uint4*)&smb[ALOFF + base + kbA + 8];
        }
#pragma unroll
        for (int ni = 0; ni < 2; ++ni) {
            const int n = waveN * 32 + ni * 16 + mrow;
            const int base = n * 40 + kbB;
            // lanes 0-15: K=0..15 ; lanes 16-31: K=16..31 (contiguous 32B)
            bH[ni].q[0] = *(const uint4*)&smb[BHOFF + base];
            bH[ni].q[1] = *(const uint4*)&smb[BHOFF + base + 8];
            bL[ni].q[0] = *(const uint4*)&smb[BLOFF + base];
            bL[ni].q[1] = *(const uint4*)&smb[BLOFF + base + 8];
        }
#pragma unroll
        for (int mi = 0; mi < 2; ++mi)
#pragma unroll
            for (int ni = 0; ni < 2; ++ni) {
                acc[mi][ni] = __builtin_amdgcn_wmma_f32_16x16x32_bf16(
                    false, aH[mi].v, false, bH[ni].v, (short)0, acc[mi][ni], false, false);
                acc[mi][ni] = __builtin_amdgcn_wmma_f32_16x16x32_bf16(
                    false, aH[mi].v, false, bL[ni].v, (short)0, acc[mi][ni], false, false);
                acc[mi][ni] = __builtin_amdgcn_wmma_f32_16x16x32_bf16(
                    false, aL[mi].v, false, bH[ni].v, (short)0, acc[mi][ni], false, false);
            }
    };

#if HAVE_ASYNC_LDS
    // Async DMA stage of one K-chunk into LDS buffer `buf` (6 B128 per thread).
    auto stage = [&](int kc, int buf) {
        unsigned short* smb = sm + buf * SMHALF;
#pragma unroll
        for (int i = 0; i < 2; ++i) {                 // A tile: 128 rows x 32 k
            const int pos = t + i * 256;
            const int r = pos >> 2, c = pos & 3;      // c: 16B chunk in row
            const size_t ga = (size_t)(m0 + r) * ENGH + (size_t)kc * 32 + c * 8;
            async_cp16(fhi + ga, smb + AHOFF + r * 40 + c * 8);
            async_cp16(flo + ga, smb + ALOFF + r * 40 + c * 8);
        }
        {                                             // B tile: 64 rows x 32 k
            const int r = t >> 2, c = t & 3;
            const size_t gb = wbase + (size_t)(n0 + r) * ENGH + (size_t)kc * 32 + c * 8;
            async_cp16(whi + gb, smb + BHOFF + r * 40 + c * 8);
            async_cp16(wlo + gb, smb + BLOFF + r * 40 + c * 8);
        }
    };

    stage(0, 0);
    __builtin_amdgcn_s_wait_asynccnt(0);
    __syncthreads();
    int cur = 0;
    for (int kc = 0; kc < ENGH / 32; ++kc) {
        if (kc + 1 < ENGH / 32) stage(kc + 1, cur ^ 1);   // DMA next chunk
        compute(sm + cur * SMHALF);                       // WMMA current chunk
        __builtin_amdgcn_s_wait_asynccnt(0);              // next buffer landed
        __syncthreads();                                  // everyone done reading cur
        cur ^= 1;
    }
#else
    // Fallback: synchronous staging through VGPRs, single buffer.
    for (int kc = 0; kc < ENGH / 32; ++kc) {
        __syncthreads();
#pragma unroll
        for (int i = 0; i < 4; ++i) {
            const int pos = t + i * 256;
            const int r = pos >> 3, c = pos & 7;
            const size_t ga = (size_t)(m0 + r) * ENGH + (size_t)kc * 32 + c * 4;
            *(uint2*)&sm[AHOFF + r * 40 + c * 4] = *(const uint2*)(fhi + ga);
            *(uint2*)&sm[ALOFF + r * 40 + c * 4] = *(const uint2*)(flo + ga);
        }
#pragma unroll
        for (int i = 0; i < 2; ++i) {
            const int pos = t + i * 256;
            const int r = pos >> 3, c = pos & 7;
            const size_t gb = wbase + (size_t)(n0 + r) * ENGH + (size_t)kc * 32 + c * 4;
            *(uint2*)&sm[BHOFF + r * 40 + c * 4] = *(const uint2*)(whi + gb);
            *(uint2*)&sm[BLOFF + r * 40 + c * 4] = *(const uint2*)(wlo + gb);
        }
        __syncthreads();
        compute(sm);
    }
#endif

    // ---- epilogue: bias add + store (C/D layout: VGPR j -> row j / j+8)
    const float* bias = (g == 0) ? value_b : (key_b + (size_t)(g - 1) * HIDN);
#pragma unroll
    for (int ni = 0; ni < 2; ++ni) {
        const int ncol = n0 + waveN * 32 + ni * 16 + mrow;
        const float bv = bias[ncol];
#pragma unroll
        for (int mi = 0; mi < 2; ++mi) {
            const int mbase = m0 + waveM * 32 + mi * 16 + hi16 * 8;
            float* op = out + ((size_t)g * MM + mbase) * HIDN + ncol;
#pragma unroll
            for (int j = 0; j < 8; ++j)
                op[(size_t)j * HIDN] = acc[mi][ni][j] + bv;
        }
    }
}

// ---------------------------------------------------------------------------
// Kernel 4: in-place RMS-norm over the last dim for the 4 key matrices.
// 1 block per (c, row); 256 threads x 4 elements.
// ---------------------------------------------------------------------------
__global__ void rmsnorm_kernel(float* __restrict__ out,
                               const float* __restrict__ norm_w) {
    const int blk = blockIdx.x;          // 0 .. 4*MM-1
    const int c = blk >> 13;             // / MM
    const int m = blk & (MM - 1);
    float* row = out + ((size_t)(1 + c) * MM + m) * HIDN;
    const int t = threadIdx.x;

    float x[4];
    float s = 0.f;
#pragma unroll
    for (int i = 0; i < 4; ++i) {
        x[i] = row[t + i * 256];
        s += x[i] * x[i];
    }
    __shared__ float red[256];
    red[t] = s;
    __syncthreads();
    for (int o = 128; o > 0; o >>= 1) {
        if (t < o) red[t] += red[t + o];
        __syncthreads();
    }
    const float scale = rsqrtf(red[0] * (1.0f / HIDN) + 1.1920928955078125e-07f);
#pragma unroll
    for (int i = 0; i < 4; ++i)
        row[t + i * 256] = x[i] * scale * norm_w[(size_t)c * HIDN + t + i * 256];
}

// ---------------------------------------------------------------------------
extern "C" void kernel_launch(void* const* d_in, const int* in_sizes, int n_in,
                              void* d_out, int out_size, void* d_ws, size_t ws_size,
                              hipStream_t stream) {
    (void)in_sizes; (void)n_in; (void)out_size; (void)ws_size;

    const int*   ids  = (const int*)d_in[0];     // input_ids (B,S)
    const int*   mult = (const int*)d_in[1];     // multipliers (3)
    const int*   vs   = (const int*)d_in[2];     // vocab_sizes (16)
    const int*   off  = (const int*)d_in[3];     // offsets (16)
    const float* emb  = (const float*)d_in[4];   // emb_table (total_N, 64)
    const float* vw   = (const float*)d_in[5];   // value_w (1024,1024)
    const float* vb   = (const float*)d_in[6];   // value_b (1024)
    const float* kw   = (const float*)d_in[7];   // key_w (4,1024,1024)
    const float* kb   = (const float*)d_in[8];   // key_b (4,1024)
    const float* nw   = (const float*)d_in[9];   // norm_w (4,1024)
    float* out = (float*)d_out;                  // (5, 8192, 1024) f32

    // Workspace layout (bf16 planes), total ~54.5 MB:
    unsigned short* fhi = (unsigned short*)d_ws;                       // 8192*1024
    unsigned short* flo = fhi + (size_t)MM * ENGH;                     // 8192*1024
    unsigned short* whi = flo + (size_t)MM * ENGH;                     // 5*1024*1024
    unsigned short* wlo = whi + (size_t)NMAT * HIDN * ENGH;            // 5*1024*1024

    gather_feat_kernel<<<MM, 256, 0, stream>>>(ids, mult, vs, off, emb, fhi, flo);
    wconv_kernel<<<(NMAT * HIDN * ENGH) / (256 * 4), 256, 0, stream>>>(vw, kw, whi, wlo);
    gemm_bf16x3_kernel<<<dim3((NMAT * HIDN) / 64, MM / 128), 256, 0, stream>>>(
        fhi, flo, whi, wlo, vb, kb, out);
    rmsnorm_kernel<<<4 * MM, 256, 0, stream>>>(out, nw);
}
